// BehlerG3_51049981280467
// MI455X (gfx1250) — compile-verified, matrix operand order checked by reading, not run on hardware
//
#include <hip/hip_runtime.h>
#include <hip/hip_bf16.h>

typedef __attribute__((ext_vector_type(16))) _Float16 v16h;
typedef __attribute__((ext_vector_type(8)))  _Float16 v8h;
typedef __attribute__((ext_vector_type(8)))  float    v8f;

#define T_DIM 512
#define WPB   8   // waves per block

__device__ __forceinline__ float cutfn(float r) {
    // Behler cosine cutoff, rc = 6: 0.5*(cos(pi*r/6)+1) for r < 6 else 0
    float c = 0.5f * (__cosf(r * 0.52359877559829887308f) + 1.0f);
    return (r < 6.0f) ? c : 0.0f;
}

__global__ __launch_bounds__(256) void behler_g3_wmma(
    const float* __restrict__ r_ij, const float* __restrict__ r_ik,
    const float* __restrict__ r_jk, const int* __restrict__ mask,
    const float* __restrict__ etas, float* __restrict__ out, int npairs)
{
    // Per-wave private tiles. Atile[.][e][t] (16x32 f16, row-major = WMMA A
    // fragment rows). Btile stored transposed: Btile[.][n][k] so B-fragment
    // loads are contiguous in K, mirroring the A layout.
    __shared__ __align__(16) _Float16 Atile[WPB][16][32];
    __shared__ __align__(16) _Float16 Btile[WPB][16][32];

    const int lane = threadIdx.x & 31;
    const int wv   = threadIdx.x >> 5;
    const int pair = blockIdx.x * WPB + wv;
    if (pair >= npairs) return;              // wave-uniform guard

    const long row0 = (long)pair * 2;        // this wave handles rows row0, row0+1

    float et[8];
    #pragma unroll
    for (int e = 0; e < 8; ++e) et[e] = etas[e];   // uniform scalar loads

    v8f acc = {};   // 16x16 f32 accumulator (diag 8x8 blocks are the results)

    for (int chunk = 0; chunk < 16; ++chunk) {
        const int t = chunk * 32 + lane;     // triple index within T
        #pragma unroll
        for (int sel = 0; sel < 2; ++sel) {  // sel=0 -> n0 (rows/cols 0-7), sel=1 -> n1
            const long idx = (row0 + sel) * T_DIM + t;

            float rij = r_ij[idx];
            float rik = r_ik[idx];
            float rjk = r_jk[idx];
            int   m   = mask[idx];

            // prefetch next K-chunk of this stream (speculative, OOB-safe)
            __builtin_prefetch(r_ij + idx + 32, 0, 0);
            __builtin_prefetch(r_ik + idx + 32, 0, 0);
            __builtin_prefetch(r_jk + idx + 32, 0, 0);
            __builtin_prefetch(mask + idx + 32, 0, 0);

            float r2 = rij * rij + rik * rik;
            float fc = cutfn(rij) * cutfn(rik);
            float ct = (r2 - rjk * rjk) / (2.0f * rij * rik);
            float u  = 0.5f * (1.0f - ct);               // in [0,1] for valid triangles
            u = fminf(fmaxf(u, 0.0f), 1.0f);
            if (m == 0) { fc = 0.0f; u = 0.0f; }         // zeros both radius and ang

            // Radial features: exp(-eta*(rij^2+rik^2)) * fc(rij)*fc(rik) in [0,1] -> f16 safe
            #pragma unroll
            for (int e = 0; e < 8; ++e) {
                float rad = __expf(-et[e] * r2) * fc;
                Atile[wv][sel * 8 + e][lane] = (_Float16)rad;
            }

            // Angular base: store u^zeta in [0,1] (zeta = 1,2,4,16 via squaring);
            // exact power-of-two prefactors folded into the f32 output scale.
            float u2 = u * u, u4 = u2 * u2, u8 = u4 * u4, u16 = u8 * u8;
            float up[4] = { u, u2, u4, u16 };
            #pragma unroll
            for (int a = 0; a < 4; ++a) {
                _Float16 h = (_Float16)up[a];
                Btile[wv][sel * 8 + a    ][lane] = h;    // 2^(1-z) branch column
                Btile[wv][sel * 8 + a + 4][lane] = h;    // 2^(1+z) branch column
            }
        }

        // Build WMMA fragments per the 16-bit 16x32 lane layout:
        //   lanes 0-15 : row/col = lane,    halves 0-7 = K0-7,  8-15 = K16-23
        //   lanes 16-31: row/col = lane-16, halves 0-7 = K8-15, 8-15 = K24-31
        const int r16  = lane & 15;
        const int koff = (lane >> 4) << 3;   // 0 or 8
        union { v16h v; v8h h[2]; } ua, ub;
        ua.h[0] = *(const v8h*)&Atile[wv][r16][koff];
        ua.h[1] = *(const v8h*)&Atile[wv][r16][koff + 16];
        ub.h[0] = *(const v8h*)&Btile[wv][r16][koff];
        ub.h[1] = *(const v8h*)&Btile[wv][r16][koff + 16];

        acc = __builtin_amdgcn_wmma_f32_16x16x32_f16(
            /*neg_a=*/false, ua.v, /*neg_b=*/false, ub.v,
            /*c_mod=*/(short)0, acc, /*reuse_a=*/false, /*reuse_b=*/false);
    }

    // Per-column scale, branchless: cols 0-3 -> 2 ; cols 4-7 -> 2^(1+2z), z={1,2,4,16}.
    // Pack the 8 exponents {1,1,1,1,3,5,9,33} as bytes of one 64-bit literal and
    // build the f32 directly in the exponent field (values are exact powers of 2).
    const int a = lane & 7;
    const unsigned long long expTab = 0x2109050301010101ull;
    const unsigned ebits = (unsigned)((expTab >> (a * 8)) & 0xffull);
    const float scale = __uint_as_float((127u + ebits) << 23);

    // D layout: VGPR i = row i (lanes 0-15, col=lane) / row 8+i (lanes 16-31, col=lane-16).
    // Useful blocks: n0 -> rows 0-7 x cols 0-7 (lanes 0-7); n1 -> rows 8-15 x cols 8-15 (lanes 24-31).
    const bool lo_blk = (lane < 8);
    const bool hi_blk = (lane >= 24);
    if (lo_blk || hi_blk) {
        const long nrow = lo_blk ? row0 : (row0 + 1);
        float* dst = out + nrow * 64 + a;      // out[(nrow*8 + e)*8 + a]
        #pragma unroll
        for (int e = 0; e < 8; ++e)
            dst[e * 8] = acc[e] * scale;
    }
}

extern "C" void kernel_launch(void* const* d_in, const int* in_sizes, int n_in,
                              void* d_out, int out_size, void* d_ws, size_t ws_size,
                              hipStream_t stream) {
    const float* r_ij = (const float*)d_in[0];
    const float* r_ik = (const float*)d_in[1];
    const float* r_jk = (const float*)d_in[2];
    const int*   mask = (const int*)d_in[3];
    const float* etas = (const float*)d_in[4];
    float*       out  = (float*)d_out;

    const int rows   = in_sizes[0] / T_DIM;          // B*N = 4096
    const int npairs = rows / 2;                     // 2048 waves of work
    const int blocks = (npairs + WPB - 1) / WPB;     // 8 waves / 256 threads per block

    behler_g3_wmma<<<blocks, 256, 0, stream>>>(r_ij, r_ik, r_jk, mask, etas, out, npairs);
}
